// VectorQuantizer_1348619730948
// MI455X (gfx1250) — matrix operand order repaired and checked by reading, not compiled, not run on hardware
//
#include <hip/hip_runtime.h>
#include <hip/hip_bf16.h>

// Problem constants (from reference): B=32, C=64, H=64, W=64, K=1024
#define C_DIM    64
#define K_CODES  1024
#define HW       4096                    // H*W
#define N_ROWS   (32 * HW)               // B*H*W = 131072
#define N_PAIRS  (N_ROWS / 32)           // 4096 pairs of 16-row tiles
#define OUT_ELEMS (32 * C_DIM * HW)      // 8388608
#define LOSS_SCALE (1.25f / 8388608.0f)  // (codebook + 0.25*commitment) / mean

typedef __attribute__((ext_vector_type(16))) _Float16 v16h;
typedef __attribute__((ext_vector_type(8)))  float    v8f;

union AFrag {           // 16 f16 elements = 8 VGPRs (WMMA A/B operand)
    v16h      v;
    _Float16  h[16];
    uint4     q[2];
};

__global__ void vq_init_loss(float* __restrict__ loss) {
    if (threadIdx.x == 0 && blockIdx.x == 0) loss[0] = 0.0f;
}

__global__ __launch_bounds__(256)
void vq_main(const float* __restrict__ z_e,
             const float* __restrict__ cb,
             float* __restrict__ out,
             float* __restrict__ loss)
{
    extern __shared__ unsigned char smem[];
    _Float16* s_cb   = (_Float16*)smem;                      // [K*C] f16, 128 KB
    float*    s_norm = (float*)(smem + K_CODES * C_DIM * 2); // [K] f32,   4 KB

    const int tid = threadIdx.x;

    // ---- Stage codebook into LDS as f16; compute ||e_k||^2 in f32 ----
    for (int i = tid; i < K_CODES * C_DIM; i += 256)
        s_cb[i] = (_Float16)cb[i];
    for (int k = tid; k < K_CODES; k += 256) {
        float s = 0.0f;
        #pragma unroll
        for (int c = 0; c < C_DIM; ++c) { float v = cb[k * C_DIM + c]; s += v * v; }
        s_norm[k] = s;
    }
    __syncthreads();

    const int lane = tid & 31;
    const int half = lane >> 4;      // 0: lanes 0-15, 1: lanes 16-31
    const int lrow = lane & 15;
    const int gwave  = blockIdx.x * 8 + (tid >> 5);
    const int nwaves = gridDim.x * 8;

    float lsum = 0.0f;

    for (int pair = gwave; pair < N_PAIRS; pair += nwaves) {
        const int n0  = pair * 32;       // 32 rows per wave iteration (2 M-tiles)
        const int b   = n0 >> 12;        // n0 / HW  (32 | HW so both tiles share b)
        const int hw0 = n0 & (HW - 1);
        const float* zb = z_e + (size_t)b * C_DIM * HW + hw0;  // index: c*HW + row
        float*       ob = out + (size_t)b * C_DIM * HW + hw0;

        // Prefetch next pair's z region (gfx1250 global_prefetch_b8)
        if (pair + nwaves < N_PAIRS) {
            const int nn = (pair + nwaves) * 32;
            const float* pz = z_e + (size_t)(nn >> 12) * C_DIM * HW + (nn & (HW - 1))
                              + (size_t)(half * 8 + (lrow & 7)) * HW + lrow;
            __builtin_prefetch(pz, 0, 0);
        }

        // ---- A fragments for both 16-row sub-tiles (rows lrow and 16+lrow) ----
        // 16-bit A 16x32 layout (contiguous, matches B): lane row=lrow holds K = half*16 + 0..15
        AFrag a0L, a1L, a0H, a1H;
        #pragma unroll
        for (int e = 0; e < 16; ++e) {
            const int c = half * 16 + e;     // frag0: c in 0..31
            a0L.h[e] = (_Float16)zb[(size_t)c * HW + lrow];
            a1L.h[e] = (_Float16)zb[(size_t)(c + 32) * HW + lrow];
            a0H.h[e] = (_Float16)zb[(size_t)c * HW + 16 + lrow];
            a1H.h[e] = (_Float16)zb[(size_t)(c + 32) * HW + 16 + lrow];
        }

        float minv[2][8];
        int   mini[2][8];
        #pragma unroll
        for (int s = 0; s < 2; ++s)
            #pragma unroll
            for (int r = 0; r < 8; ++r) { minv[s][r] = 3.4e38f; mini[s][r] = 0; }

        // B fragment fetch from LDS: col = lrow, lane-half selects K 0..15 / 16..31
        auto loadB = [&](AFrag& c0, AFrag& c1, int t) {
            const int code = t * 16 + lrow;
            const uint4* p = (const uint4*)(s_cb + (size_t)code * C_DIM + half * 16);
            c0.q[0] = p[0];  c0.q[1] = p[1];   // c = half*16 + 0..15
            c1.q[0] = p[4];  c1.q[1] = p[5];   // c = 32 + half*16 + 0..15
        };

        // Two independent WMMA accumulator chains (L and H sub-tiles) hide the
        // WMMA->VALU hazard window; argmin update is fully branchless.
        auto compute = [&](const AFrag& b0, const AFrag& b1, int t) {
            const int code = t * 16 + lrow;
            v8f accL = {0.f, 0.f, 0.f, 0.f, 0.f, 0.f, 0.f, 0.f};
            v8f accH = {0.f, 0.f, 0.f, 0.f, 0.f, 0.f, 0.f, 0.f};
            accL = __builtin_amdgcn_wmma_f32_16x16x32_f16(
                       false, a0L.v, false, b0.v, (short)0, accL, false, false);
            accH = __builtin_amdgcn_wmma_f32_16x16x32_f16(
                       false, a0H.v, false, b0.v, (short)0, accH, false, false);
            accL = __builtin_amdgcn_wmma_f32_16x16x32_f16(
                       false, a1L.v, false, b1.v, (short)0, accL, false, false);
            accH = __builtin_amdgcn_wmma_f32_16x16x32_f16(
                       false, a1H.v, false, b1.v, (short)0, accH, false, false);
            const float nrm = s_norm[code];
            #pragma unroll
            for (int r = 0; r < 8; ++r) {
                // distance (minus row-constant ||z||^2) = ||e||^2 - 2 z.e
                const float cL = __builtin_fmaf(-2.0f, accL[r], nrm);
                const float cH = __builtin_fmaf(-2.0f, accH[r], nrm);
                const bool tL = cL < minv[0][r];
                const bool tH = cH < minv[1][r];
                minv[0][r] = tL ? cL : minv[0][r];
                mini[0][r] = tL ? code : mini[0][r];
                minv[1][r] = tH ? cH : minv[1][r];
                mini[1][r] = tH ? code : mini[1][r];
            }
        };

        // ---- Sweep 64 code tiles, ping-pong double-buffered B fragments ----
        AFrag pA0, pA1, pB0, pB1;
        loadB(pA0, pA1, 0);
        for (int t = 0; t < 64; t += 2) {
            loadB(pB0, pB1, t + 1);            // issue next loads before WMMAs
            compute(pA0, pA1, t);
            loadB(pA0, pA1, (t + 2) & 63);     // wraps harmlessly at t=62
            compute(pB0, pB1, t + 1);
        }

        // ---- Argmin across the 16 lanes of each half (wave32), branchless ----
        #pragma unroll
        for (int m = 1; m < 16; m <<= 1) {
            #pragma unroll
            for (int s = 0; s < 2; ++s) {
                #pragma unroll
                for (int r = 0; r < 8; ++r) {
                    const float ov = __shfl_xor(minv[s][r], m, 32);
                    const int   oi = __shfl_xor(mini[s][r], m, 32);
                    const bool take = (ov < minv[s][r]) |
                                      ((ov == minv[s][r]) & (oi < mini[s][r]));
                    minv[s][r] = take ? ov : minv[s][r];
                    mini[s][r] = take ? oi : mini[s][r];
                }
            }
        }

        // ---- Gather codes, scatter z_q (B,C,H,W), accumulate loss ----
        // lane handles row = s*16 + r + half*8 (its own mini[s][r]), 4 ch per r
        #pragma unroll
        for (int s = 0; s < 2; ++s) {
            #pragma unroll
            for (int r = 0; r < 8; ++r) {
                const int row = s * 16 + r + half * 8;
                const int idx = mini[s][r];
                const float* crow = cb + (size_t)idx * C_DIM;
                #pragma unroll
                for (int cc = 0; cc < 4; ++cc) {
                    const int c = lrow + cc * 16;
                    const float qv = crow[c];
                    const size_t off = (size_t)c * HW + row;
                    const float d = qv - zb[off];
                    ob[off] = qv;              // z_q_st forward value == z_q
                    lsum += d * d;
                }
            }
        }
    }

    // ---- Wave-reduce loss, single atomic per wave ----
    #pragma unroll
    for (int m = 1; m < 32; m <<= 1) lsum += __shfl_xor(lsum, m, 32);
    if (lane == 0) atomicAdd(loss, lsum * LOSS_SCALE);
}

extern "C" void kernel_launch(void* const* d_in, const int* in_sizes, int n_in,
                              void* d_out, int out_size, void* d_ws, size_t ws_size,
                              hipStream_t stream) {
    const float* z_e = (const float*)d_in[0];   // (32,64,64,64) f32
    const float* cb  = (const float*)d_in[1];   // (1024,64) f32
    float* out  = (float*)d_out;                // 8388608 z_q + 1 loss
    float* loss = out + OUT_ELEMS;

    vq_init_loss<<<1, 32, 0, stream>>>(loss);

    const size_t smem_bytes = (size_t)K_CODES * C_DIM * 2 + (size_t)K_CODES * 4; // 135168
    vq_main<<<dim3(256), dim3(256), smem_bytes, stream>>>(z_e, cb, out, loss);
}
